// Encoder_45079976739359
// MI455X (gfx1250) — matrix operand order repaired and checked by reading, not compiled, not run on hardware
//
#include <hip/hip_runtime.h>
#include <hip/hip_bf16.h>

// ---- problem constants (from reference) ----
#define D     512
#define MROWS 64
#define KTOP  5
#define TBANK 30000
#define WGRP  1536           // NKEEP / MROWS (pad == 0)
#define CBLK  1875           // TBANK / 16
#define NSLICE 25            // column slices for parallelism
#define CBPER  75            // CBLK / NSLICE
#define EPSN  1e-8f

typedef __attribute__((ext_vector_type(16))) _Float16 v16h;
typedef __attribute__((ext_vector_type(8)))  _Float16 v8h;
typedef __attribute__((ext_vector_type(8)))  float    v8f;
typedef __attribute__((ext_vector_type(4)))  float    v4f;
typedef __attribute__((ext_vector_type(2)))  float    v2f;

// ---------------------------------------------------------------------------
// Kernel 1: group-average 64 rows -> normalize -> f16  (1536 blocks x 128 thr)
// ---------------------------------------------------------------------------
__global__ __launch_bounds__(128)
void group_avg_norm_k(const float* __restrict__ x, _Float16* __restrict__ a16) {
    const int g = blockIdx.x, t = threadIdx.x;
    const int col = t * 4;
    const float* base = x + (size_t)g * MROWS * D + col;
    v4f s = {0.f, 0.f, 0.f, 0.f};
    #pragma unroll 8
    for (int r = 0; r < MROWS; ++r)
        s += *(const v4f*)(base + (size_t)r * D);
    s *= (1.0f / (float)MROWS);
    float p = s.x * s.x + s.y * s.y + s.z * s.z + s.w * s.w;
    __shared__ float red[128];
    red[t] = p; __syncthreads();
    for (int off = 64; off > 0; off >>= 1) {
        if (t < off) red[t] += red[t + off];
        __syncthreads();
    }
    const float inv = 1.0f / fmaxf(sqrtf(red[0]), EPSN);
    _Float16* o = a16 + (size_t)g * D + col;
    o[0] = (_Float16)(s.x * inv); o[1] = (_Float16)(s.y * inv);
    o[2] = (_Float16)(s.z * inv); o[3] = (_Float16)(s.w * inv);
}

// ---------------------------------------------------------------------------
// Kernel 2: row-normalize short_text -> f16   (30000 blocks x 128 thr)
// ---------------------------------------------------------------------------
__global__ __launch_bounds__(128)
void norm_text_k(const float* __restrict__ st, _Float16* __restrict__ t16) {
    const int row = blockIdx.x, t = threadIdx.x;
    const int col = t * 4;
    v4f s = *(const v4f*)(st + (size_t)row * D + col);
    float p = s.x * s.x + s.y * s.y + s.z * s.z + s.w * s.w;
    __shared__ float red[128];
    red[t] = p; __syncthreads();
    for (int off = 64; off > 0; off >>= 1) {
        if (t < off) red[t] += red[t + off];
        __syncthreads();
    }
    const float inv = 1.0f / fmaxf(sqrtf(red[0]), EPSN);
    _Float16* o = t16 + (size_t)row * D + col;
    o[0] = (_Float16)(s.x * inv); o[1] = (_Float16)(s.y * inv);
    o[2] = (_Float16)(s.z * inv); o[3] = (_Float16)(s.w * inv);
}

// ---------------------------------------------------------------------------
// Kernel 3: cos = a16 @ t16^T via v_wmma_f32_16x16x32_f16, per-slice top-5.
// 300 blocks x 8 waves = 2400 waves: wave = (row_block 0..95, slice 0..24).
// Waves in a block share the slice -> identical B streams (L0 broadcast).
// A (16x512 f16) register-resident; B double-buffered across k-steps so each
// WMMA waits on the PREVIOUS iteration's loads (s_wait_loadcnt <= 2).
// ---------------------------------------------------------------------------
__global__ __launch_bounds__(256)
void cos_topk_k(const _Float16* __restrict__ a16,
                const _Float16* __restrict__ t16,
                float* __restrict__ pv, int* __restrict__ pi) {
    __shared__ float tile[8][256];          // 16x16 f32 tile per wave
    const int wid  = threadIdx.x >> 5;
    const int lane = threadIdx.x & 31;
    const int wg   = blockIdx.x * 8 + wid;  // 0..2399
    const int rb   = wg % 96;               // row block
    const int slc  = wg / 96;               // column slice
    const int m    = lane & 15;
    const int half = lane >> 4;
    float* mytile = &tile[wid][0];

    // A fragments: 16 k-steps, register resident.
    union U { v16h v; v8h h[2]; };
    const _Float16* arow = a16 + (size_t)(rb * 16 + m) * D;
    U Af[16];
    #pragma unroll
    for (int ks = 0; ks < 16; ++ks) {
        const int k0 = ks * 32;
        Af[ks].h[0] = *(const v8h*)(arow + k0 + half * 16);
        Af[ks].h[1] = *(const v8h*)(arow + k0 + half * 16 + 8);
    }

    float t5v[KTOP]; int t5i[KTOP];
    #pragma unroll
    for (int j = 0; j < KTOP; ++j) { t5v[j] = -1e30f; t5i[j] = 0; }

    const int cb0 = slc * CBPER;
    // preload B fragment for (cb0, ks=0)
    U Bc, Bn;
    {
        const _Float16* br = t16 + (size_t)(cb0 * 16 + m) * D;
        Bc.h[0] = *(const v8h*)(br + half * 16);
        Bc.h[1] = *(const v8h*)(br + half * 16 + 8);
    }

    for (int i = 0; i < CBPER; ++i) {
        const int cb = cb0 + i;
        const _Float16* brow  = t16 + (size_t)(cb * 16 + m) * D;
        const int cbn = (i + 1 < CBPER) ? (cb + 1) : cb;   // clamp prefetch
        const _Float16* brown = t16 + (size_t)(cbn * 16 + m) * D;

        v8f c = {0.f, 0.f, 0.f, 0.f, 0.f, 0.f, 0.f, 0.f};
        #pragma unroll
        for (int ks = 0; ks < 16; ++ks) {
            // prefetch next fragment (next k-step, or next cb's ks=0)
            if (ks < 15) {
                const int k1 = (ks + 1) * 32;
                Bn.h[0] = *(const v8h*)(brow + k1 + half * 16);
                Bn.h[1] = *(const v8h*)(brow + k1 + half * 16 + 8);
            } else {
                Bn.h[0] = *(const v8h*)(brown + half * 16);
                Bn.h[1] = *(const v8h*)(brown + half * 16 + 8);
            }
            c = __builtin_amdgcn_wmma_f32_16x16x32_f16(
                    false, Af[ks].v, false, Bc.v, (short)0, c, false, false);
            Bc = Bn;
        }

        // C/D layout: VGPR r -> (M = r + 8*half, N = lane&15)
        #pragma unroll
        for (int r = 0; r < 8; ++r)
            mytile[(r + 8 * half) * 16 + m] = c[r];
        __asm__ volatile("s_wait_dscnt 0" ::: "memory");   // wave-local LDS RAW

        if (lane < 16) {                     // lane = local row
            #pragma unroll
            for (int j = 0; j < 16; ++j) {
                const float v = mytile[lane * 16 + j];
                if (v > t5v[KTOP - 1]) {     // strict '>' keeps stable tie order
                    t5v[KTOP - 1] = v; t5i[KTOP - 1] = cb * 16 + j;
                    #pragma unroll
                    for (int p = KTOP - 1; p > 0; --p) {
                        if (t5v[p] > t5v[p - 1]) {
                            float tv = t5v[p]; t5v[p] = t5v[p-1]; t5v[p-1] = tv;
                            int   ti = t5i[p]; t5i[p] = t5i[p-1]; t5i[p-1] = ti;
                        }
                    }
                }
            }
        }
    }
    if (lane < 16) {
        const int row = rb * 16 + lane;
        const size_t o = ((size_t)row * NSLICE + slc) * KTOP;
        #pragma unroll
        for (int j = 0; j < KTOP; ++j) { pv[o + j] = t5v[j]; pi[o + j] = t5i[j]; }
    }
}

// ---------------------------------------------------------------------------
// Kernel 3b: merge per-slice top-5 candidates -> final top-5 per row.
// Ascending slice order == ascending column index (stable ties via strict '>').
// ---------------------------------------------------------------------------
__global__ __launch_bounds__(256)
void merge_topk_k(const float* __restrict__ pv, const int* __restrict__ pi,
                  int* __restrict__ topi) {
    const int row = blockIdx.x * 256 + threadIdx.x;
    if (row >= WGRP) return;
    float t5v[KTOP]; int t5i[KTOP];
    #pragma unroll
    for (int j = 0; j < KTOP; ++j) { t5v[j] = -1e30f; t5i[j] = 0; }
    const size_t base = (size_t)row * NSLICE * KTOP;
    for (int s = 0; s < NSLICE; ++s) {
        #pragma unroll
        for (int j = 0; j < KTOP; ++j) {
            const float v = pv[base + s * KTOP + j];
            const int   id = pi[base + s * KTOP + j];
            if (v > t5v[KTOP - 1]) {
                t5v[KTOP - 1] = v; t5i[KTOP - 1] = id;
                #pragma unroll
                for (int p = KTOP - 1; p > 0; --p) {
                    if (t5v[p] > t5v[p - 1]) {
                        float tv = t5v[p]; t5v[p] = t5v[p-1]; t5v[p-1] = tv;
                        int   ti = t5i[p]; t5i[p] = t5i[p-1]; t5i[p-1] = ti;
                    }
                }
            }
        }
    }
    #pragma unroll
    for (int j = 0; j < KTOP; ++j) topi[row * KTOP + j] = t5i[j];
}

// ---------------------------------------------------------------------------
// Kernel 4: topk_avg[i] = mean of the 5 selected short_text rows (fp32 exact)
// ---------------------------------------------------------------------------
__global__ __launch_bounds__(128)
void gather_avg_k(const float* __restrict__ st, const int* __restrict__ topi,
                  float* __restrict__ avg) {
    const int row = blockIdx.x, t = threadIdx.x;
    const int col = t * 4;
    v4f s = {0.f, 0.f, 0.f, 0.f};
    #pragma unroll
    for (int j = 0; j < KTOP; ++j) {
        const int ix = topi[row * KTOP + j];
        s += *(const v4f*)(st + (size_t)ix * D + col);
    }
    s *= (1.0f / (float)KTOP);
    *(v4f*)(avg + (size_t)row * D + col) = s;
}

// ---------------------------------------------------------------------------
// Kernel 5: out = topk_avg @ W^T + b via fp32-exact v_wmma_f32_16x16x4_f32.
// 3072 16x16 tiles -> 384 blocks x 8 waves. Bias folded into accumulator.
// ---------------------------------------------------------------------------
__global__ __launch_bounds__(256)
void out_gemm_k(const float* __restrict__ avg, const float* __restrict__ W,
                const float* __restrict__ bias, float* __restrict__ out) {
    const int wid  = threadIdx.x >> 5;
    const int lane = threadIdx.x & 31;
    const int wg   = blockIdx.x * 8 + wid;   // tile id 0..3071
    const int tm   = wg >> 5;                // 0..95
    const int tn   = wg & 31;                // 0..31
    const int n    = lane & 15;
    const int half = lane >> 4;
    const int col  = tn * 16 + n;
    const float* arow = avg + (size_t)(tm * 16 + n) * D;  // A row m = lane&15
    const float* brow = W   + (size_t)col * D;            // B(k,n) = W[n][k]
    const float bv = bias[col];
    v8f c;
    #pragma unroll
    for (int r = 0; r < 8; ++r) c[r] = bv;
    #pragma unroll 8
    for (int ks = 0; ks < D / 4; ++ks) {
        const int k0 = ks * 4 + half * 2;    // A 16x4 f32: K = half*2 + {0,1}
        v2f A = *(const v2f*)(arow + k0);
        v2f B = *(const v2f*)(brow + k0);
        c = __builtin_amdgcn_wmma_f32_16x16x4_f32(
                false, A, false, B, (short)0, c, false, false);
    }
    #pragma unroll
    for (int r = 0; r < 8; ++r)
        out[(size_t)(tm * 16 + r + 8 * half) * D + col] = c[r];
}

// ---------------------------------------------------------------------------
// Workspace layout (bytes, 256B-aligned):
//   a16:  1536*512*2      = 1,572,864   @ 0
//   t16:  30000*512*2     = 30,720,000  @ 1,572,864
//   topi: 1536*5*4        = 30,720      @ 32,292,864
//   avg:  1536*512*4      = 3,145,728   @ 32,323,584
//   pv:   1536*25*5*4     = 768,000     @ 35,469,312
//   pi:   1536*25*5*4     = 768,000     @ 36,237,312   (total ~37 MB)
// ---------------------------------------------------------------------------
extern "C" void kernel_launch(void* const* d_in, const int* in_sizes, int n_in,
                              void* d_out, int out_size, void* d_ws, size_t ws_size,
                              hipStream_t stream) {
    const float* x  = (const float*)d_in[0];   // x_plip  (1,131072,512)
    const float* st = (const float*)d_in[1];   // short_text (30000,512)
    const float* W  = (const float*)d_in[2];   // W_down (512,512)
    const float* b  = (const float*)d_in[3];   // b_down (512)
    float* out = (float*)d_out;                // (1536,512)

    char* ws = (char*)d_ws;
    _Float16* a16  = (_Float16*)(ws + 0);
    _Float16* t16  = (_Float16*)(ws + 1572864);
    int*      topi = (int*)     (ws + 32292864);
    float*    avg  = (float*)   (ws + 32323584);
    float*    pv   = (float*)   (ws + 35469312);
    int*      pi   = (int*)     (ws + 36237312);

    group_avg_norm_k<<<WGRP, 128, 0, stream>>>(x, a16);
    norm_text_k     <<<TBANK, 128, 0, stream>>>(st, t16);
    cos_topk_k      <<<300, 256, 0, stream>>>(a16, t16, pv, pi);
    merge_topk_k    <<<6, 256, 0, stream>>>(pv, pi, topi);
    gather_avg_k    <<<WGRP, 128, 0, stream>>>(st, topi, avg);
    out_gemm_k      <<<384, 256, 0, stream>>>(avg, W, b, out);
}
